// CalculateAttention_39453569581284
// MI455X (gfx1250) — matrix operand order repaired and checked by reading, not compiled, not run on hardware
//
#include <hip/hip_runtime.h>
#include <hip/hip_bf16.h>

typedef __attribute__((ext_vector_type(16))) _Float16 v16h;
typedef __attribute__((ext_vector_type(8)))  _Float16 v8h;
typedef __attribute__((ext_vector_type(4)))  _Float16 v4h;
typedef __attribute__((ext_vector_type(8)))  float    v8f;

#define B_   4
#define H_   16
#define S_   2048
#define D_   64
#define KT   64            // keys per LDS tile
#define WAVES 8
#define QROWS 32           // query rows per wave (two 16-row A tiles)
#define QB   (QROWS*WAVES) // 256 query rows per workgroup
#define LDK  72            // LDS row stride in halves (pad to dodge bank conflicts)
#define LDM  80            // mask LDS row stride in bytes (16B aligned)

// Fold 1/sqrt(D) * log2(e) into Q so scores come out of the QK^T WMMA already
// in the exp2 domain: p = exp2(s - max). Identical math, one v_exp_f32 per exp.
#define QSCALE   0.18033688011112042f          // 0.125 * log2(e)
#define MASKFILL (-1.8033688011112042e8f)      // -1e9 * QSCALE

#if __has_builtin(__builtin_amdgcn_exp2f)
#define EXP2F(x) __builtin_amdgcn_exp2f(x)     // v_exp_f32
#else
#define EXP2F(x) exp2f(x)
#endif

__global__ __launch_bounds__(256)
void fa_fwd_f16wmma(const float* __restrict__ Q,
                    const float* __restrict__ K,
                    const float* __restrict__ V,
                    const unsigned char* __restrict__ mask,
                    float* __restrict__ out)
{
    __shared__ _Float16 Kl[KT * LDK];             // [key][d], row-major f16
    __shared__ _Float16 Vt[D_ * LDK];             // [d][key], transposed f16
    __shared__ _Float16 Pl[WAVES * QROWS * LDK];  // per-wave P scratch [m][key]
    __shared__ unsigned char Ml[WAVES * QROWS * LDM]; // per-wave mask tile [m][key]

    const int tid  = threadIdx.x;
    const int lane = tid & 31;
    const int wave = tid >> 5;
    const int n    = lane & 15;        // N / column index in B- and C-layouts
    const int hi   = (lane >> 4) & 1;  // half-wave select
    const int qb   = blockIdx.x;
    const int h    = blockIdx.y;
    const int b    = blockIdx.z;
    const int q0   = qb * QB + wave * QROWS;

    const size_t headOff = ((size_t)(b * H_ + h)) * S_ * D_;
    const float* Qh = Q + headOff;
    const float* Kh = K + headOff;
    const float* Vh = V + headOff;
    float*       Oh = out + headOff;
    const unsigned char* Mh = mask + (size_t)b * S_ * S_;

    // ---- Q A-fragments: [qt][f], 16x32 f16 each (contiguous k = 16*hi + s order),
    // pre-scaled by QSCALE so WMMA emits exp2-domain scores directly.
    v16h aq[2][2];
    #pragma unroll
    for (int qt = 0; qt < 2; ++qt) {
        const float* qr = Qh + (size_t)(q0 + 16 * qt + n) * D_;
        #pragma unroll
        for (int f = 0; f < 2; ++f) {
            const int d0 = 32 * f + 16 * hi;       // halves 0..7  -> K d0..d0+7
            const int d1 = 32 * f + 16 * hi + 8;   // halves 8..15 -> K d1..d1+7
            #pragma unroll
            for (int i = 0; i < 8; ++i) aq[qt][f][i]     = (_Float16)(qr[d0 + i] * QSCALE);
            #pragma unroll
            for (int i = 0; i < 8; ++i) aq[qt][f][8 + i] = (_Float16)(qr[d1 + i] * QSCALE);
        }
    }

    v8f acc[2][4];                    // per qt: 16x64 output as 4 C-tiles
    float mrow[2][8], lrow[2][8];
    #pragma unroll
    for (int qt = 0; qt < 2; ++qt) {
        #pragma unroll
        for (int t = 0; t < 4; ++t) acc[qt][t] = {};
        #pragma unroll
        for (int r = 0; r < 8; ++r) { mrow[qt][r] = -1e30f; lrow[qt][r] = 0.0f; }
    }

    _Float16*      Pw = &Pl[wave * QROWS * LDK];
    unsigned char* Mw = &Ml[wave * QROWS * LDM];

    for (int k0 = 0; k0 < S_; k0 += KT) {
        __syncthreads();  // previous iteration's LDS readers done

        // ---- cooperative stage: K tile (row-major) + V tile (transposed), f32 -> f16
        {
            const int row = tid >> 2;         // 0..63 : key within tile
            const int dg  = (tid & 3) * 16;   // 0,16,32,48 : d group
            const float* kr = Kh + (size_t)(k0 + row) * D_ + dg;
            const float* vr = Vh + (size_t)(k0 + row) * D_ + dg;
            #pragma unroll
            for (int i = 0; i < 4; ++i) {
                float4 kv = *(const float4*)(kr + 4 * i);
                v4h hk; hk[0] = (_Float16)kv.x; hk[1] = (_Float16)kv.y;
                        hk[2] = (_Float16)kv.z; hk[3] = (_Float16)kv.w;
                *(v4h*)&Kl[row * LDK + dg + 4 * i] = hk;
                float4 vv = *(const float4*)(vr + 4 * i);
                Vt[(dg + 4 * i + 0) * LDK + row] = (_Float16)vv.x;
                Vt[(dg + 4 * i + 1) * LDK + row] = (_Float16)vv.y;
                Vt[(dg + 4 * i + 2) * LDK + row] = (_Float16)vv.z;
                Vt[(dg + 4 * i + 3) * LDK + row] = (_Float16)vv.w;
            }
            // mask tile for this wave's 32 q rows: coalesced 16B loads
            const unsigned char* mg = Mh + (size_t)(q0 + lane) * S_ + k0;
            #pragma unroll
            for (int i = 0; i < 4; ++i) {
                uint4 mv = *(const uint4*)(mg + 16 * i);
                *(uint4*)&Mw[lane * LDM + 16 * i] = mv;
            }
            // prefetch next tile's K/V lines into WGP cache (global_prefetch_b8)
            if (k0 + KT < S_) {
                __builtin_prefetch((const void*)(kr + (size_t)KT * D_), 0, 3);
                __builtin_prefetch((const void*)(vr + (size_t)KT * D_), 0, 3);
            }
        }
        __syncthreads();

        // ---- QK^T + mask + online softmax + P store, one q-subtile at a time
        // (keeps score-tile liveness at 4 v8f to stay under 256 VGPRs)
        #pragma unroll
        for (int qt = 0; qt < 2; ++qt) {
            v8f sc[4];
            #pragma unroll
            for (int nt = 0; nt < 4; ++nt) sc[nt] = {};
            #pragma unroll
            for (int nt = 0; nt < 4; ++nt) {
                #pragma unroll
                for (int f = 0; f < 2; ++f) {
                    const _Float16* bp = &Kl[(nt * 16 + n) * LDK + 32 * f + 16 * hi];
                    v8h b0 = *(const v8h*)bp;
                    v8h b1 = *(const v8h*)(bp + 8);
                    v16h bb = __builtin_shufflevector(b0, b1,
                        0,1,2,3,4,5,6,7,8,9,10,11,12,13,14,15);
                    sc[nt] = __builtin_amdgcn_wmma_f32_16x16x32_f16(
                        false, aq[qt][f], false, bb, (short)0, sc[nt], false, false);
                }
            }

            #pragma unroll
            for (int nt = 0; nt < 4; ++nt) {
                #pragma unroll
                for (int r = 0; r < 8; ++r) {
                    // byte is 0/1; |s| << ulp(MASKFILL) so this equals the
                    // reference's masked fill after exp2 (underflow to 0).
                    float mb = (float)Mw[(16 * qt + 8 * hi + r) * LDM + nt * 16 + n];
                    sc[nt][r] += mb * MASKFILL;
                }
            }

            #pragma unroll
            for (int r = 0; r < 8; ++r) {
                float m0 = fmaxf(fmaxf(sc[0][r], sc[1][r]),
                                 fmaxf(sc[2][r], sc[3][r]));
                #pragma unroll
                for (int off = 1; off < 16; off <<= 1)
                    m0 = fmaxf(m0, __shfl_xor(m0, off, 32));
                const float mn    = fmaxf(mrow[qt][r], m0);
                const float alpha = EXP2F(mrow[qt][r] - mn);
                mrow[qt][r] = mn;
                float p0 = EXP2F(sc[0][r] - mn);
                float p1 = EXP2F(sc[1][r] - mn);
                float p2 = EXP2F(sc[2][r] - mn);
                float p3 = EXP2F(sc[3][r] - mn);
                float ps = (p0 + p1) + (p2 + p3);
                const int m = 16 * qt + 8 * hi + r;
                Pw[m * LDK +  0 + n] = (_Float16)p0;
                Pw[m * LDK + 16 + n] = (_Float16)p1;
                Pw[m * LDK + 32 + n] = (_Float16)p2;
                Pw[m * LDK + 48 + n] = (_Float16)p3;
                #pragma unroll
                for (int off = 1; off < 16; off <<= 1)
                    ps += __shfl_xor(ps, off, 32);
                lrow[qt][r] = lrow[qt][r] * alpha + ps;
                #pragma unroll
                for (int t = 0; t < 4; ++t) acc[qt][t][r] *= alpha;
            }
        }

        // P is wave-private: LDS ops from one wave complete in order (DScnt),
        // so only a compiler reordering fence is needed -- no block barrier.
        asm volatile("" ::: "memory");

        // ---- acc += P * V : shared V B-fragments feed both q-subtiles ----
        #pragma unroll
        for (int f = 0; f < 2; ++f) {
            v16h pa[2];
            #pragma unroll
            for (int qt = 0; qt < 2; ++qt) {
                const _Float16* pp = &Pw[(16 * qt + n) * LDK + 32 * f + 16 * hi];
                v8h p0 = *(const v8h*)pp;
                v8h p1 = *(const v8h*)(pp + 8);
                pa[qt] = __builtin_shufflevector(p0, p1,
                    0,1,2,3,4,5,6,7,8,9,10,11,12,13,14,15);
            }
            #pragma unroll
            for (int dt = 0; dt < 4; ++dt) {
                const _Float16* vp = &Vt[(dt * 16 + n) * LDK + 32 * f + 16 * hi];
                v8h b0 = *(const v8h*)vp;
                v8h b1 = *(const v8h*)(vp + 8);
                v16h vb = __builtin_shufflevector(b0, b1,
                    0,1,2,3,4,5,6,7,8,9,10,11,12,13,14,15);
                #pragma unroll
                for (int qt = 0; qt < 2; ++qt)
                    acc[qt][dt] = __builtin_amdgcn_wmma_f32_16x16x32_f16(
                        false, pa[qt], false, vb, (short)0, acc[qt][dt], false, false);
            }
        }
    }

    // ---- epilogue: divide by row sum, store fp32 ----
    #pragma unroll
    for (int qt = 0; qt < 2; ++qt) {
        float* orow = Oh + (size_t)(q0 + 16 * qt + 8 * hi) * D_;
        #pragma unroll
        for (int r = 0; r < 8; ++r) {
            const float inv = 1.0f / lrow[qt][r];
            #pragma unroll
            for (int dt = 0; dt < 4; ++dt)
                orow[(size_t)r * D_ + dt * 16 + n] = acc[qt][dt][r] * inv;
        }
    }
}

extern "C" void kernel_launch(void* const* d_in, const int* in_sizes, int n_in,
                              void* d_out, int out_size, void* d_ws, size_t ws_size,
                              hipStream_t stream) {
    const float* Q = (const float*)d_in[0];
    const float* K = (const float*)d_in[1];
    const float* V = (const float*)d_in[2];
    const unsigned char* mask = (const unsigned char*)d_in[3];  // jnp.bool_: 1 byte
    float* out = (float*)d_out;

    dim3 grid(S_ / QB, H_, B_);   // (8, 16, 4)
    dim3 block(256);
    fa_fwd_f16wmma<<<grid, block, 0, stream>>>(Q, K, V, mask, out);
}
